// VectorQuantizerEMA_54683523613274
// MI455X (gfx1250) — matrix operand order repaired and checked by reading, not compiled, not run on hardware
//
#include <hip/hip_runtime.h>
#include <hip/hip_bf16.h>

typedef __attribute__((ext_vector_type(16))) _Float16 v16h;
typedef __attribute__((ext_vector_type(8)))  _Float16 v8h;
typedef __attribute__((ext_vector_type(8)))  float    v8f;
typedef int v4i __attribute__((vector_size(16)));

// Problem constants
#define CDIM   256        // channels
#define MCODES 2048       // codebook entries
#define NTOT   65536      // B*D*H*W = 8*8*32*32
#define DHW    8192       // D*H*W per batch sample
#define QELEMS 4194304    // total z elements
#define NCHUNK (MCODES / 16)

#define DECAY  0.99f
#define OMD    0.01f
#define EPSV   1e-5f

// ---- output float offsets (return-order concat) ----
#define OQ     0u
#define OLOSS  4194304u
#define OIDX   4194305u
#define ONEMB  4259841u
#define ONCS   4784129u
#define ONEA   4786177u

// ---- workspace byte offsets ----
#define WS_FLAT16   0u            // N*C f16  = 32 MB
#define WS_EMB16    33554432u     // M*C f16  = 1 MB
#define WS_ENORM    34603008u     // M f32    = 8 KB
#define WS_COUNTS   34611200u     // M f32    = 8 KB
#define WS_ESUM     34619392u     // M*C f32  = 2 MB
#define WS_IDX      36716544u     // N i32    = 256 KB
#define WS_SCAL     36978688u     // scalars: [0]=loss acc, [1]=n_total

// gfx1250 async global->LDS path (builtin confirmed present; falls back cleanly)
#if __has_builtin(__builtin_amdgcn_global_load_async_to_lds_b128)
#define USE_ASYNC_LDS 1
typedef __attribute__((address_space(1))) v4i* gas_ptr;
typedef __attribute__((address_space(3))) v4i* las_ptr;
#endif

// --------------------------------------------------------------------------
// 1) z (B,C,DHW) -> flat16 (N,C) f16, 32x32 LDS tile transpose
__global__ __launch_bounds__(256) void zt_kernel(const float* __restrict__ z,
                                                 _Float16* __restrict__ flat16) {
  __shared__ float tile[32][33];
  const int b  = blockIdx.z;
  const int n0 = blockIdx.y * 32;   // dhw tile
  const int c0 = blockIdx.x * 32;   // channel tile
  const int tx = threadIdx.x & 31;
  const int ty = threadIdx.x >> 5;  // 0..7
  const float* src = z + (size_t)b * CDIM * DHW;
#pragma unroll
  for (int i = 0; i < 4; ++i) {
    int c = c0 + ty + i * 8;
    tile[ty + i * 8][tx] = src[(size_t)c * DHW + n0 + tx];
  }
  __syncthreads();
  _Float16* dst = flat16 + (size_t)b * DHW * CDIM;
#pragma unroll
  for (int i = 0; i < 4; ++i) {
    int n = n0 + ty + i * 8;
    dst[(size_t)n * CDIM + c0 + tx] = (_Float16)tile[tx][ty + i * 8];
  }
}

// --------------------------------------------------------------------------
// 2) embedding -> f16 + per-code squared norm
__global__ __launch_bounds__(256) void emb_prep_kernel(const float* __restrict__ emb,
                                                       _Float16* __restrict__ emb16,
                                                       float* __restrict__ enorm) {
  __shared__ float red[256];
  const int m = blockIdx.x;
  const int c = threadIdx.x;
  float v = emb[(size_t)m * CDIM + c];
  emb16[(size_t)m * CDIM + c] = (_Float16)v;
  red[c] = v * v;
  __syncthreads();
#pragma unroll
  for (int s = 128; s > 0; s >>= 1) {
    if (c < s) red[c] += red[c + s];
    __syncthreads();
  }
  if (c == 0) enorm[m] = red[0];
}

// --------------------------------------------------------------------------
// 3) WMMA argmin. Each wave owns 16 rows; K=256 A-fragments resident in VGPRs.
//    B tiles double-buffered in LDS via async global->LDS (one barrier/chunk);
//    B fragments staged to registers so the 8 WMMAs issue back-to-back.
#define BPAD 264  // 16B-aligned row stride, conflict-free fragment reads
__global__ __launch_bounds__(256) void argmin_kernel(const _Float16* __restrict__ flat16,
                                                     const _Float16* __restrict__ emb16,
                                                     const float* __restrict__ enorm,
                                                     int* __restrict__ idx_buf,
                                                     float* __restrict__ idx_out) {
  const int lane   = threadIdx.x & 31;
  const int wave   = threadIdx.x >> 5;                 // 0..7
  const int n_base = blockIdx.x * 128 + wave * 16;
  const int arow   = lane & 15;                        // A-row / B-col within tile
  const int hi     = lane >> 4;                        // lane half (K offset select)

  __shared__ _Float16 bs[2][16][BPAD];
  __shared__ float en_s[MCODES];

  // A fragments: ISA 16-bit A 16x32 layout. lanes0-15: K=[0..7],[16..23]; lanes16-31: +8
  v16h afrag[8];
  {
    const _Float16* rowp = flat16 + (size_t)(n_base + arow) * CDIM;
#pragma unroll
    for (int j = 0; j < 8; ++j) {
      const int k0 = j * 32;
      union { v16h v; v8h h[2]; } u;
      u.h[0] = *(const v8h*)(rowp + k0 + hi * 8);
      u.h[1] = *(const v8h*)(rowp + k0 + 16 + hi * 8);
      afrag[j] = u.v;
    }
  }

  // cache all code norms in LDS
  for (int i = threadIdx.x; i < MCODES; i += 256) en_s[i] = enorm[i];

  // staging coordinates: 256 threads move 16 codes x 256 halves (2x16B each)
  const int sr = threadIdx.x >> 4;         // code row 0..15
  const int sk = (threadIdx.x & 15) * 16;  // half offset 0..240

#ifdef USE_ASYNC_LDS
#define STAGE_B(buf, mb)                                                            \
  do {                                                                              \
    const _Float16* gsrc = emb16 + (size_t)((mb) + sr) * CDIM + sk;                 \
    _Float16* ldst = &bs[buf][sr][sk];                                              \
    __builtin_amdgcn_global_load_async_to_lds_b128((gas_ptr)gsrc, (las_ptr)ldst, 0, 0); \
    __builtin_amdgcn_global_load_async_to_lds_b128((gas_ptr)(gsrc + 8),             \
                                                   (las_ptr)(ldst + 8), 0, 0);      \
  } while (0)
#if __has_builtin(__builtin_amdgcn_s_wait_asynccnt)
#define ASYNC_FENCE() __builtin_amdgcn_s_wait_asynccnt(0)
#else
#define ASYNC_FENCE() asm volatile("s_wait_asynccnt 0" ::: "memory")
#endif
#else
#define STAGE_B(buf, mb)                                                            \
  do {                                                                              \
    const v8h* gsrc = (const v8h*)(emb16 + (size_t)((mb) + sr) * CDIM + sk);        \
    v8h* ldst = (v8h*)&bs[buf][sr][sk];                                             \
    ldst[0] = gsrc[0];                                                              \
    ldst[1] = gsrc[1];                                                              \
  } while (0)
#define ASYNC_FENCE() ((void)0)
#endif

  STAGE_B(0, 0);   // prime the pipeline
  ASYNC_FENCE();
  __syncthreads();

  float minval[8];
  int   minidx[8];
#pragma unroll
  for (int r = 0; r < 8; ++r) { minval[r] = 3.4e38f; minidx[r] = 0; }

  for (int it = 0; it < NCHUNK; ++it) {
    const int m0  = it * 16;
    const int cur = it & 1;
    if (it + 1 < NCHUNK) STAGE_B(cur ^ 1, m0 + 16);  // overlap next-tile stage

    // pull all 8 B fragments into registers (one ds clause), then WMMA chain
    union { v16h v; v8h h[2]; } bf[8];
#pragma unroll
    for (int j = 0; j < 8; ++j) {
      // B 32x16 layout: lane col = arow; lanes0-15 K=[k0..k0+7],[k0+16..k0+23], lanes16-31 +8
      const v8h* bp = (const v8h*)&bs[cur][arow][j * 32 + hi * 8];
      bf[j].h[0] = bp[0];
      bf[j].h[1] = bp[2];
    }
    v8f acc = {};
#pragma unroll
    for (int j = 0; j < 8; ++j)
      acc = __builtin_amdgcn_wmma_f32_16x16x32_f16(false, afrag[j], false, bf[j].v,
                                                   (short)0, acc, false, false);

    // D layout: lane holds code column (m0+arow), rows hi*8 + r
    const float en   = en_s[m0 + arow];
    const int   code = m0 + arow;
#pragma unroll
    for (int r = 0; r < 8; ++r) {
      float d = fmaf(-2.0f, acc[r], en);
      if (d < minval[r]) { minval[r] = d; minidx[r] = code; }
    }

    ASYNC_FENCE();      // staged tile landed in LDS
    __syncthreads();    // safe to flip buffers
  }

  // reduce across the 16 lanes sharing each row group (xor masks stay in-half)
#pragma unroll
  for (int s = 1; s < 16; s <<= 1) {
#pragma unroll
    for (int r = 0; r < 8; ++r) {
      float ov = __shfl_xor(minval[r], s, 32);
      int   oi = __shfl_xor(minidx[r], s, 32);
      if (ov < minval[r] || (ov == minval[r] && oi < minidx[r])) {
        minval[r] = ov; minidx[r] = oi;
      }
    }
  }
  if ((lane & 15) == 0) {
    const int rbase = n_base + hi * 8;
#pragma unroll
    for (int r = 0; r < 8; ++r) {
      idx_buf[rbase + r] = minidx[r];
      idx_out[rbase + r] = (float)minidx[r];
    }
  }
}

// --------------------------------------------------------------------------
// 4) gather quantized (coalesced writes, embedding gather hits L2) + loss accum
__global__ __launch_bounds__(256) void gather_loss_kernel(const float* __restrict__ z,
                                                          const float* __restrict__ emb,
                                                          const int* __restrict__ idx_buf,
                                                          float* __restrict__ outq,
                                                          float* __restrict__ scal) {
  __shared__ float red[256];
  const size_t i   = (size_t)blockIdx.x * 256 + threadIdx.x;  // (b,c,dhw), dhw fastest
  const int    dhw = (int)(i & (DHW - 1));
  const int    bc  = (int)(i >> 13);
  const int    c   = bc & 255;
  const int    b   = bc >> 8;
  const int    n   = b * DHW + dhw;
  const int    m   = idx_buf[n];
  const float  q   = emb[(size_t)m * CDIM + c];
  const float  zv  = z[i];
  outq[i] = q;          // quantized_st value == quantized
  const float d = q - zv;
  red[threadIdx.x] = d * d;
  __syncthreads();
#pragma unroll
  for (int s = 128; s > 0; s >>= 1) {
    if (threadIdx.x < s) red[threadIdx.x] += red[threadIdx.x + s];
    __syncthreads();
  }
  if (threadIdx.x == 0) atomicAdd(&scal[0], red[0]);
}

// --------------------------------------------------------------------------
// 5) segment sums: counts + embed_sum (f32 atomics)
__global__ __launch_bounds__(256) void scatter_kernel(const float* __restrict__ z,
                                                      const int* __restrict__ idx_buf,
                                                      float* __restrict__ counts,
                                                      float* __restrict__ esum) {
  const int c   = blockIdx.y;                       // channel
  const int n   = blockIdx.x * 256 + threadIdx.x;   // row
  const int b   = n >> 13;
  const int dhw = n & (DHW - 1);
  const int m   = idx_buf[n];
  const float zv = z[((size_t)(b * CDIM + c) << 13) + dhw];
  atomicAdd(&esum[(size_t)m * CDIM + c], zv);
  if (c == 0) atomicAdd(&counts[m], 1.0f);
}

// --------------------------------------------------------------------------
// 6a) new_cluster_size, n_total reduction, loss finalize (single block)
__global__ __launch_bounds__(256) void ema1_kernel(const float* __restrict__ cluster_size,
                                                   const float* __restrict__ counts,
                                                   float* __restrict__ ncs_out,
                                                   float* __restrict__ scal,
                                                   float* __restrict__ loss_out) {
  __shared__ float red[256];
  float s = 0.f;
  for (int i = threadIdx.x; i < MCODES; i += 256) {
    float v = cluster_size[i] * DECAY + OMD * counts[i];
    ncs_out[i] = v;
    s += v;
  }
  red[threadIdx.x] = s;
  __syncthreads();
#pragma unroll
  for (int st = 128; st > 0; st >>= 1) {
    if (threadIdx.x < st) red[threadIdx.x] += red[threadIdx.x + st];
    __syncthreads();
  }
  if (threadIdx.x == 0) {
    scal[1] = red[0];
    loss_out[0] = scal[0] * (1.0f / (float)QELEMS);
  }
}

// 6b) new_embedding_avg + normalized new_embedding
__global__ __launch_bounds__(256) void ema2_kernel(const float* __restrict__ emb_avg,
                                                   const float* __restrict__ esum,
                                                   const float* __restrict__ ncs,
                                                   const float* __restrict__ scal,
                                                   float* __restrict__ nea_out,
                                                   float* __restrict__ nemb_out) {
  const int m = blockIdx.x;
  const int c = threadIdx.x;
  const size_t o = (size_t)m * CDIM + c;
  const float nea  = emb_avg[o] * DECAY + OMD * esum[o];
  nea_out[o] = nea;
  const float ntot = scal[1];
  const float cs   = (ncs[m] + EPSV) / (ntot + (float)MCODES * EPSV) * ntot;
  nemb_out[o] = nea / cs;
}

// --------------------------------------------------------------------------
extern "C" void kernel_launch(void* const* d_in, const int* in_sizes, int n_in,
                              void* d_out, int out_size, void* d_ws, size_t ws_size,
                              hipStream_t stream) {
  const float* z        = (const float*)d_in[0];
  const float* emb      = (const float*)d_in[1];
  const float* csz      = (const float*)d_in[2];
  const float* emb_avg  = (const float*)d_in[3];
  float* out = (float*)d_out;

  char* ws = (char*)d_ws;
  _Float16* flat16 = (_Float16*)(ws + WS_FLAT16);
  _Float16* emb16  = (_Float16*)(ws + WS_EMB16);
  float*    enorm  = (float*)(ws + WS_ENORM);
  float*    counts = (float*)(ws + WS_COUNTS);
  float*    esum   = (float*)(ws + WS_ESUM);
  int*      idxb   = (int*)(ws + WS_IDX);
  float*    scal   = (float*)(ws + WS_SCAL);

  // zero atomics/accumulators each launch (graph-capture safe)
  (void)hipMemsetAsync(counts, 0, (MCODES + MCODES * CDIM) * sizeof(float), stream); // counts+esum contiguous
  (void)hipMemsetAsync(scal, 0, 2 * sizeof(float), stream);

  zt_kernel<<<dim3(CDIM / 32, DHW / 32, 8), 256, 0, stream>>>(z, flat16);
  emb_prep_kernel<<<MCODES, 256, 0, stream>>>(emb, emb16, enorm);
  argmin_kernel<<<NTOT / 128, 256, 0, stream>>>(flat16, emb16, enorm, idxb, out + OIDX);
  gather_loss_kernel<<<QELEMS / 256, 256, 0, stream>>>(z, emb, idxb, out + OQ, scal);
  scatter_kernel<<<dim3(NTOT / 256, CDIM), 256, 0, stream>>>(z, idxb, counts, esum);
  ema1_kernel<<<1, 256, 0, stream>>>(csz, counts, out + ONCS, scal, out + OLOSS);
  ema2_kernel<<<MCODES, 256, 0, stream>>>(emb_avg, esum, out + ONCS, scal,
                                          out + ONEA, out + ONEMB);
}
